// GraphTransformerLayer_87462714015772
// MI455X (gfx1250) — matrix operand order repaired and checked by reading, not compiled, not run on hardware
//
#include <hip/hip_runtime.h>

typedef _Float16 f16;
typedef __attribute__((ext_vector_type(8)))  _Float16 v8h;
typedef __attribute__((ext_vector_type(16))) _Float16 v16h;
typedef __attribute__((ext_vector_type(8)))  float    v8f;

#define CDIM   256
#define HDIM   512
#define NN     256
#define BB     4
#define HEADS  8
#define MROWS  64
#define XSTR   264   // f16 LDS stride for 256-wide activations
#define HSTR   528   // f16 LDS stride for 512-wide hidden
#define YSTR   260   // f32 LDS stride
#define NT     256
#define LNEPS  1e-5f

// ---------------- WMMA GEMM core ----------------------------------------
// A-fragment (16x32 f16): lanes 0-15 hold K[0..15], lanes 16-31 hold K[16..31],
// 16 contiguous halves loaded as two 8-half chunks.
__device__ inline v16h make_afrag(const f16* X, int xs, int m, int k0) {
  const v8h lo = *(const v8h*)(X + m * xs + k0);
  const v8h hi = *(const v8h*)(X + m * xs + k0 + 8);
  return __builtin_shufflevector(lo, hi, 0,1,2,3,4,5,6,7,8,9,10,11,12,13,14,15);
}

// Compute 4 n-tiles (nt0, nt0+2, nt0+4, nt0+6) for m-tile `mt` of a 64-row LDS
// activation X against transposed f16 weight WT[N][K]. B-fragment: lane n =
// nt*16 + (lane&15), 16 contiguous K-halves at (lane>>4)*16.
template<int K>
__device__ inline void gemm4(const f16* X, int xs, const f16* __restrict__ WT,
                             int nt0, int mt, int lane, v8f acc[4]) {
  const int mr = mt * 16 + (lane & 15);
  const int ks = lane >> 4;
  for (int ko = 0; ko < K / 32; ++ko) {
    const v16h a  = make_afrag(X, xs, mr, ko * 32 + ks * 16);
    const int  kb = ko * 32 + ks * 16;
#pragma unroll
    for (int t = 0; t < 4; ++t) {
      const int n = (nt0 + 2 * t) * 16 + (lane & 15);
      const v16h b = *(const v16h*)(WT + (long)n * K + kb);
      acc[t] = __builtin_amdgcn_wmma_f32_16x16x32_f16(
          false, a, false, b, (short)0, acc[t], false, false);
    }
  }
}

// ---------------- row LayerNorm over 64x256 f32 in LDS -------------------
__device__ inline void row_ln(float* sY, const float* __restrict__ g,
                              const float* __restrict__ b,
                              f16* xdst,                 // if non-null: write f16 copy
                              float* gout, long gbase,   // else: write f32 to global
                              float* red, float* stat, int tid) {
  __syncthreads();
  {
    const int r = tid >> 2, p = tid & 3;
    float s = 0.f, s2 = 0.f;
    const float* row = sY + r * YSTR + p * 64;
    for (int c = 0; c < 64; ++c) { const float v = row[c]; s += v; s2 += v * v; }
    red[(r * 4 + p) * 2]     = s;
    red[(r * 4 + p) * 2 + 1] = s2;
  }
  __syncthreads();
  if (tid < MROWS) {
    float s = 0.f, s2 = 0.f;
    for (int p = 0; p < 4; ++p) { s += red[(tid*4+p)*2]; s2 += red[(tid*4+p)*2+1]; }
    const float mu  = s  * (1.f / CDIM);
    const float var = s2 * (1.f / CDIM) - mu * mu;
    stat[tid * 2]     = mu;
    stat[tid * 2 + 1] = rsqrtf(var + LNEPS);
  }
  __syncthreads();
  for (int idx = tid; idx < MROWS * CDIM; idx += NT) {
    const int r = idx >> 8, c = idx & 255;
    const float v = (sY[r * YSTR + c] - stat[r * 2]) * stat[r * 2 + 1] * g[c] + b[c];
    if (xdst) { sY[r * YSTR + c] = v; xdst[r * XSTR + c] = (f16)v; }
    else      { gout[gbase + (long)r * CDIM + c] = v; }
  }
  __syncthreads();
}

// ---------------- projection + bias + residual -> sY ---------------------
__device__ inline void proj_residual(const f16* sAtt, const f16* __restrict__ WT,
                                     const float* __restrict__ bo,
                                     const float* __restrict__ res, long R0,
                                     float* sY, int wv, int lane) {
  const int mt = wv >> 1, nh = wv & 1;
  for (int gg = 0; gg < 2; ++gg) {
    v8f acc[4] = {};
    gemm4<CDIM>(sAtt, XSTR, WT, gg * 8 + nh, mt, lane, acc);
    const int rr0 = mt * 16 + (lane >> 4) * 8;
#pragma unroll
    for (int t = 0; t < 4; ++t) {
      const int c  = (gg * 8 + nh + 2 * t) * 16 + (lane & 15);
      const float bc = bo[c];
#pragma unroll
      for (int v = 0; v < 8; ++v) {
        const int m = rr0 + v;
        sY[m * YSTR + c] = acc[t][v] + bc + res[(R0 + m) * (long)CDIM + c];
      }
    }
  }
}

// ---------------- FFN: relu(X@W1+b1)@W2+b2 added into sY -----------------
__device__ inline void ffn_tail(const f16* sX, f16* sH, float* sY,
                                const f16* __restrict__ W1T, const float* __restrict__ b1,
                                const f16* __restrict__ W2T, const float* __restrict__ b2,
                                int wv, int lane) {
  const int mt = wv >> 1, nh = wv & 1;
  for (int gg = 0; gg < 4; ++gg) {            // hidden = 512 cols
    v8f acc[4] = {};
    gemm4<CDIM>(sX, XSTR, W1T, gg * 8 + nh, mt, lane, acc);
    const int rr0 = mt * 16 + (lane >> 4) * 8;
#pragma unroll
    for (int t = 0; t < 4; ++t) {
      const int c  = (gg * 8 + nh + 2 * t) * 16 + (lane & 15);
      const float bc = b1[c];
#pragma unroll
      for (int v = 0; v < 8; ++v) {
        const float u = acc[t][v] + bc;
        sH[(rr0 + v) * HSTR + c] = (f16)(u > 0.f ? u : 0.f);
      }
    }
  }
  __syncthreads();
  for (int gg = 0; gg < 2; ++gg) {            // back to 256 cols
    v8f acc[4] = {};
    gemm4<HDIM>(sH, HSTR, W2T, gg * 8 + nh, mt, lane, acc);
    const int rr0 = mt * 16 + (lane >> 4) * 8;
#pragma unroll
    for (int t = 0; t < 4; ++t) {
      const int c  = (gg * 8 + nh + 2 * t) * 16 + (lane & 15);
      const float bc = b2[c];
#pragma unroll
      for (int v = 0; v < 8; ++v) {
        const int m = rr0 + v;
        sY[m * YSTR + c] += acc[t][v] + bc;
      }
    }
  }
  __syncthreads();
}

// ---------------- fused edge kernel --------------------------------------
__global__ __launch_bounds__(256) void edge_kernel(
    const float* __restrict__ e,
    const f16* __restrict__ WeT,  const f16* __restrict__ WoeT,
    const float* __restrict__ boe,
    const f16* __restrict__ We1T, const float* __restrict__ be1,
    const f16* __restrict__ We2T, const float* __restrict__ be2,
    const float* __restrict__ g1e, const float* __restrict__ b1e,
    const float* __restrict__ g2e, const float* __restrict__ b2e,
    const float* __restrict__ Qf,  const float* __restrict__ Kf,
    float* __restrict__ Sout, float* __restrict__ out_e) {
  extern __shared__ char smem[];
  f16*   sX   = (f16*)smem;                    // 64 x 264 f16
  f16*   sH   = (f16*)(smem + 33792);          // 64 x 528 f16 (att / hidden)
  float* sY   = (float*)(smem + 101376);       // 64 x 260 f32
  float* sS   = (float*)(smem + 167936);       // 64 x 8 head sums
  float* red  = (float*)(smem + 169984);
  float* stat = (float*)(smem + 172032);

  const int tid = threadIdx.x, wv = tid >> 5, lane = tid & 31;
  const long R0 = (long)blockIdx.x * MROWS;    // 64 edge rows share (b, i)
  const int b  = (int)(R0 / ((long)NN * NN));
  const int i  = (int)((R0 / NN) % NN);
  const int j0 = (int)(R0 % NN);

  for (int idx = tid; idx < MROWS * HEADS; idx += NT) sS[idx] = 0.f;
  { // load e rows -> f16 LDS
    const float4* ep = (const float4*)(e + R0 * CDIM);
    for (int idx = tid; idx < MROWS * CDIM / 4; idx += NT) {
      const float4 vv = ep[idx];
      const int r = idx >> 6, c = (idx & 63) * 4;
      f16* d = sX + r * XSTR + c;
      d[0] = (f16)vv.x; d[1] = (f16)vv.y; d[2] = (f16)vv.z; d[3] = (f16)vv.w;
    }
  }
  __syncthreads();

  // GEMM1: Pe = e@We, gated by Q_i*K_j/sqrt(d); emit att (f16) + head sums
  {
    const int mt = wv >> 1, nh = wv & 1;
    const float isd = 0.17677669529663688f;    // 1/sqrt(32)
    const long qb = ((long)(b * NN + i)) * CDIM;
    for (int gg = 0; gg < 2; ++gg) {
      v8f acc[4] = {};
      gemm4<CDIM>(sX, XSTR, WeT, gg * 8 + nh, mt, lane, acc);
      const int rr0 = mt * 16 + (lane >> 4) * 8;
#pragma unroll
      for (int t = 0; t < 4; ++t) {
        const int c  = (gg * 8 + nh + 2 * t) * 16 + (lane & 15);
        const float qs = Qf[qb + c] * isd;
        const int hh = c >> 5;
#pragma unroll
        for (int v = 0; v < 8; ++v) {
          const int m = rr0 + v;
          const float kv = Kf[((long)(b * NN + j0 + m)) * CDIM + c];
          const float sc = acc[t][v] * qs * kv;
          atomicAdd(&sS[m * HEADS + hh], sc);
          sH[m * XSTR + c] = (f16)sc;          // e_attn tile, stride XSTR
        }
      }
    }
  }
  __syncthreads();

  // clipped per-head sums -> workspace for the softmax kernel
  for (int idx = tid; idx < MROWS * HEADS; idx += NT) {
    const int m = idx >> 3, hh = idx & 7;
    const float v = fminf(5.f, fmaxf(-5.f, sS[idx]));
    Sout[(((long)(b * HEADS + hh) * NN + i)) * NN + (j0 + m)] = v;
  }

  proj_residual(sH, WoeT, boe, e, R0, sY, wv, lane);          // e_attn@Woe + e
  row_ln(sY, g1e, b1e, sX, nullptr, 0, red, stat, tid);       // e1 (f32+f16)
  ffn_tail(sX, sH, sY, We1T, be1, We2T, be2, wv, lane);       // + FFN(e1)
  row_ln(sY, g2e, b2e, nullptr, out_e, R0 * CDIM, red, stat, tid); // e2 out
}

// ---------------- node pipeline kernel -----------------------------------
__global__ __launch_bounds__(256) void node_kernel(
    const float* __restrict__ h, const float* __restrict__ Hattn,
    const f16* __restrict__ WohT, const float* __restrict__ boh,
    const f16* __restrict__ Wh1T, const float* __restrict__ bh1,
    const f16* __restrict__ Wh2T, const float* __restrict__ bh2,
    const float* __restrict__ g1h, const float* __restrict__ b1h,
    const float* __restrict__ g2h, const float* __restrict__ b2h,
    float* __restrict__ out_h) {
  extern __shared__ char smem[];
  f16*   sX   = (f16*)smem;
  f16*   sH   = (f16*)(smem + 33792);
  float* sY   = (float*)(smem + 101376);
  float* red  = (float*)(smem + 169984);
  float* stat = (float*)(smem + 172032);

  const int tid = threadIdx.x, wv = tid >> 5, lane = tid & 31;
  const long R0 = (long)blockIdx.x * MROWS;
  { // load h_attn rows -> f16 LDS (att input)
    const float4* ap = (const float4*)(Hattn + R0 * CDIM);
    for (int idx = tid; idx < MROWS * CDIM / 4; idx += NT) {
      const float4 vv = ap[idx];
      const int r = idx >> 6, c = (idx & 63) * 4;
      f16* d = sH + r * XSTR + c;
      d[0] = (f16)vv.x; d[1] = (f16)vv.y; d[2] = (f16)vv.z; d[3] = (f16)vv.w;
    }
  }
  __syncthreads();
  proj_residual(sH, WohT, boh, h, R0, sY, wv, lane);
  row_ln(sY, g1h, b1h, sX, nullptr, 0, red, stat, tid);
  ffn_tail(sX, sH, sY, Wh1T, bh1, Wh2T, bh2, wv, lane);
  row_ln(sY, g2h, b2h, nullptr, out_h, R0 * CDIM, red, stat, tid);
}

// ---------------- Q/K/V projections --------------------------------------
__global__ __launch_bounds__(256) void qkv_kernel(
    const float* __restrict__ h,
    const f16* __restrict__ WqT, const f16* __restrict__ WkT,
    const f16* __restrict__ WvT,
    float* __restrict__ Qf, float* __restrict__ Kf, float* __restrict__ Vf) {
  __shared__ f16 sX[MROWS * XSTR];
  const int tid = threadIdx.x, wv = tid >> 5, lane = tid & 31;
  const long R0 = (long)blockIdx.x * MROWS;
  {
    const float4* hp = (const float4*)(h + R0 * CDIM);
    for (int idx = tid; idx < MROWS * CDIM / 4; idx += NT) {
      const float4 vv = hp[idx];
      const int r = idx >> 6, c = (idx & 63) * 4;
      f16* d = sX + r * XSTR + c;
      d[0] = (f16)vv.x; d[1] = (f16)vv.y; d[2] = (f16)vv.z; d[3] = (f16)vv.w;
    }
  }
  __syncthreads();
  const int mt = wv >> 1, nh = wv & 1;
  const f16* Ws[3] = { WqT, WkT, WvT };
  float*     Os[3] = { Qf, Kf, Vf };
  for (int s = 0; s < 3; ++s) {
    for (int gg = 0; gg < 2; ++gg) {
      v8f acc[4] = {};
      gemm4<CDIM>(sX, XSTR, Ws[s], gg * 8 + nh, mt, lane, acc);
      const int rr0 = mt * 16 + (lane >> 4) * 8;
#pragma unroll
      for (int t = 0; t < 4; ++t) {
        const int c = (gg * 8 + nh + 2 * t) * 16 + (lane & 15);
#pragma unroll
        for (int v = 0; v < 8; ++v)
          Os[s][(R0 + rr0 + v) * (long)CDIM + c] = acc[t][v];
      }
    }
  }
}

// ---------------- softmax over j + h_attn = w @ V ------------------------
__global__ __launch_bounds__(256) void attn_kernel(
    const float* __restrict__ S, const float* __restrict__ Vf,
    float* __restrict__ Hattn) {
  __shared__ float wbuf[HEADS][NN];
  const int tid = threadIdx.x, wv = tid >> 5, lane = tid & 31;
  const int b = blockIdx.x >> 8, i = blockIdx.x & 255;

  const float* srow = S + ((long)(b * HEADS + wv) * NN + i) * NN;
  float vals[8];
  float mx = -1e30f;
#pragma unroll
  for (int q = 0; q < 8; ++q) { vals[q] = srow[lane + q * 32]; mx = fmaxf(mx, vals[q]); }
  for (int d = 16; d >= 1; d >>= 1) mx = fmaxf(mx, __shfl_xor(mx, d, 32));
  float sum = 0.f;
#pragma unroll
  for (int q = 0; q < 8; ++q) { vals[q] = __expf(vals[q] - mx); sum += vals[q]; }
  for (int d = 16; d >= 1; d >>= 1) sum += __shfl_xor(sum, d, 32);
  const float inv = 1.f / sum;
#pragma unroll
  for (int q = 0; q < 8; ++q) wbuf[wv][lane + q * 32] = vals[q] * inv;
  __syncthreads();

  const int c = tid, hh = c >> 5;
  const float* vb = Vf + (long)(b * NN) * CDIM + c;
  float acc = 0.f;
  for (int j = 0; j < NN; ++j) acc += wbuf[hh][j] * vb[(long)j * CDIM];
  Hattn[((long)(b * NN) + i) * CDIM + c] = acc;
}

// ---------------- weight f32 -> f16 transpose ----------------------------
__global__ void wprep_kernel(const float* __restrict__ src, f16* __restrict__ dst,
                             int K, int Nout) {
  const int idx = blockIdx.x * blockDim.x + threadIdx.x;
  if (idx >= K * Nout) return;
  const int k = idx / Nout, n = idx % Nout;
  dst[(long)n * K + k] = (f16)src[idx];
}

// ---------------- host launcher ------------------------------------------
extern "C" void kernel_launch(void* const* d_in, const int* in_sizes, int n_in,
                              void* d_out, int out_size, void* d_ws, size_t ws_size,
                              hipStream_t stream) {
  const float* h   = (const float*)d_in[0];
  const float* e   = (const float*)d_in[1];
  const float* Wq  = (const float*)d_in[2];
  const float* Wk  = (const float*)d_in[3];
  const float* Wv  = (const float*)d_in[4];
  const float* We  = (const float*)d_in[5];
  const float* Woh = (const float*)d_in[6];
  const float* boh = (const float*)d_in[7];
  const float* Woe = (const float*)d_in[8];
  const float* boe = (const float*)d_in[9];
  const float* g1h = (const float*)d_in[10];
  const float* b1h = (const float*)d_in[11];
  const float* g1e = (const float*)d_in[12];
  const float* b1e = (const float*)d_in[13];
  const float* g2h = (const float*)d_in[14];
  const float* b2h = (const float*)d_in[15];
  const float* g2e = (const float*)d_in[16];
  const float* b2e = (const float*)d_in[17];
  const float* Wh1 = (const float*)d_in[18];
  const float* bh1 = (const float*)d_in[19];
  const float* Wh2 = (const float*)d_in[20];
  const float* bh2 = (const float*)d_in[21];
  const float* We1 = (const float*)d_in[22];
  const float* be1 = (const float*)d_in[23];
  const float* We2 = (const float*)d_in[24];
  const float* be2 = (const float*)d_in[25];

  float* out_h = (float*)d_out;
  float* out_e = out_h + (long)BB * NN * CDIM;

  char* ws = (char*)d_ws;
  f16* WqT  = (f16*)ws;
  f16* WkT  = WqT  + 65536;
  f16* WvT  = WkT  + 65536;
  f16* WeT  = WvT  + 65536;
  f16* WohT = WeT  + 65536;
  f16* WoeT = WohT + 65536;
  f16* Wh1T = WoeT + 65536;
  f16* Wh2T = Wh1T + 131072;
  f16* We1T = Wh2T + 131072;
  f16* We2T = We1T + 131072;
  float* Qf    = (float*)(ws + 1835008);
  float* Kf    = Qf + (long)BB * NN * CDIM;
  float* Vf    = Kf + (long)BB * NN * CDIM;
  float* Hattn = Vf + (long)BB * NN * CDIM;
  float* Sbuf  = Hattn + (long)BB * NN * CDIM;   // B*H*N*N floats

  auto wp = [&](const float* s, f16* d, int K, int Nout) {
    const int tot = K * Nout;
    wprep_kernel<<<(tot + 255) / 256, 256, 0, stream>>>(s, d, K, Nout);
  };
  wp(Wq, WqT, CDIM, CDIM);   wp(Wk, WkT, CDIM, CDIM);
  wp(Wv, WvT, CDIM, CDIM);   wp(We, WeT, CDIM, CDIM);
  wp(Woh, WohT, CDIM, CDIM); wp(Woe, WoeT, CDIM, CDIM);
  wp(Wh1, Wh1T, CDIM, HDIM); wp(Wh2, Wh2T, HDIM, CDIM);
  wp(We1, We1T, CDIM, HDIM); wp(We2, We2T, HDIM, CDIM);

  qkv_kernel<<<BB * NN / MROWS, NT, 0, stream>>>(h, WqT, WkT, WvT, Qf, Kf, Vf);

  const size_t shmem = 172544;   // < 320 KB WGP LDS
  (void)hipFuncSetAttribute((const void*)edge_kernel,
      hipFuncAttributeMaxDynamicSharedMemorySize, (int)shmem);
  (void)hipFuncSetAttribute((const void*)node_kernel,
      hipFuncAttributeMaxDynamicSharedMemorySize, (int)shmem);

  edge_kernel<<<BB * NN * NN / MROWS, NT, shmem, stream>>>(
      e, WeT, WoeT, boe, We1T, be1, We2T, be2,
      g1e, b1e, g2e, b2e, Qf, Kf, Sbuf, out_e);

  attn_kernel<<<BB * NN, NT, 0, stream>>>(Sbuf, Vf, Hattn);

  node_kernel<<<BB * NN / MROWS, NT, shmem, stream>>>(
      h, Hattn, WohT, boh, Wh1T, bh1, Wh2T, bh2,
      g1h, b1h, g2h, b2h, out_h);
}
